// SelfAttention_80900003987672
// MI455X (gfx1250) — compile-verified, matrix operand order checked by reading, not run on hardware
//
#include <hip/hip_runtime.h>
#include <math.h>

typedef __bf16 bf16_t;
typedef __attribute__((ext_vector_type(16))) __bf16 v16bf;
typedef __attribute__((ext_vector_type(8)))  __bf16 v8bf;
typedef __attribute__((ext_vector_type(8)))  float  v8f;

__device__ __forceinline__ bf16_t f2bf(float x) {
  unsigned u = __float_as_uint(x);
  unsigned r = (u + 0x7FFFu + ((u >> 16) & 1u)) >> 16;   // round-to-nearest-even
  return __builtin_bit_cast(bf16_t, (unsigned short)r);
}
__device__ __forceinline__ unsigned short f2bf_bits(float x) {
  unsigned u = __float_as_uint(x);
  return (unsigned short)((u + 0x7FFFu + ((u >> 16) & 1u)) >> 16);
}

// ---------------- fp32 -> bf16 conversion (vectorized: 16B in, 8B out) ----------------
__global__ void cvt_f32_bf16(const float4* __restrict__ src,
                             ushort4* __restrict__ dst, int n4) {
  int i = blockIdx.x * blockDim.x + threadIdx.x;
  int stride = gridDim.x * blockDim.x;
  for (; i < n4; i += stride) {
    float4 v = src[i];
    ushort4 o;
    o.x = f2bf_bits(v.x); o.y = f2bf_bits(v.y);
    o.z = f2bf_bits(v.z); o.w = f2bf_bits(v.w);
    dst[i] = o;
  }
}

// ---------------- generic bf16 NT GEMM: D = scale*(A @ B^T) + bias ----------------
// A: [M,K] row-major, B: [N,K] row-major. All shapes compile-time.
// Block tile 128x128x32; 256 threads = 8 wave32 waves (4Mx2N), wave = 32x64 = 2x4 WMMA tiles.
// Tile staging: CDNA5 async global->LDS (ASYNCcnt) with double-buffered LDS.
#define BM 128
#define BN 128
#define BK 32
#define LDSROW 40   // 32+8 pad bf16; 80B row stride: 16B aligned, conflict-free fragment reads

template <int LDA, int LDB, int KTOT, long long SAZ, long long SBZ,
          bool HAS_BIAS, bool OUTF32, int LDC, long long SCZ,
          bool SPLIT, int SPLITN, long long PLANE>
__global__ __launch_bounds__(256, 2)
void gemm_bf16_nt(const bf16_t* __restrict__ A,
                  const bf16_t* __restrict__ B,
                  float* __restrict__ Cf,
                  bf16_t* __restrict__ Cb,
                  const float* __restrict__ bias,
                  float scale) {
  __shared__ bf16_t As[2][BM * LDSROW];
  __shared__ bf16_t Bs[2][BN * LDSROW];

  const int tid  = threadIdx.x;
  const int lane = tid & 31;
  const int wave = tid >> 5;
  const int wm   = wave >> 1;          // 0..3
  const int wn   = wave & 1;           // 0..1
  const long long m0 = (long long)blockIdx.y * BM;
  const long long n0 = (long long)blockIdx.x * BN;

  A += (long long)blockIdx.z * SAZ;
  B += (long long)blockIdx.z * SBZ;

  // Per-thread staging geometry: 2 A-chunks + 2 B-chunks of 16B per iteration.
  const int row0 = tid >> 2;                 // chunk tid
  const int col0 = (tid & 3) << 3;
  const int row1 = (tid + 256) >> 2;         // chunk tid+256
  const int col1 = ((tid + 256) & 3) << 3;

  // Issue one tile's async loads (4 wave-wide async ops per wave).
  auto issue = [&](int buf, int k0) {
    unsigned la0 = (unsigned)(uintptr_t)&As[buf][row0 * LDSROW + col0];
    unsigned la1 = (unsigned)(uintptr_t)&As[buf][row1 * LDSROW + col1];
    unsigned lb0 = (unsigned)(uintptr_t)&Bs[buf][row0 * LDSROW + col0];
    unsigned lb1 = (unsigned)(uintptr_t)&Bs[buf][row1 * LDSROW + col1];
    unsigned long long ga0 = (unsigned long long)(uintptr_t)(A + (m0 + row0) * LDA + k0 + col0);
    unsigned long long ga1 = (unsigned long long)(uintptr_t)(A + (m0 + row1) * LDA + k0 + col1);
    unsigned long long gb0 = (unsigned long long)(uintptr_t)(B + (n0 + row0) * LDB + k0 + col0);
    unsigned long long gb1 = (unsigned long long)(uintptr_t)(B + (n0 + row1) * LDB + k0 + col1);
    asm volatile("global_load_async_to_lds_b128 %0, %1, off" :: "v"(la0), "v"(ga0) : "memory");
    asm volatile("global_load_async_to_lds_b128 %0, %1, off" :: "v"(la1), "v"(ga1) : "memory");
    asm volatile("global_load_async_to_lds_b128 %0, %1, off" :: "v"(lb0), "v"(gb0) : "memory");
    asm volatile("global_load_async_to_lds_b128 %0, %1, off" :: "v"(lb1), "v"(gb1) : "memory");
  };

  v8f acc[2][4];
#pragma unroll
  for (int i = 0; i < 2; ++i)
#pragma unroll
    for (int j = 0; j < 4; ++j)
#pragma unroll
      for (int r = 0; r < 8; ++r) acc[i][j][r] = 0.0f;

  // Per-lane fragment geometry (16-bit A 16x32 layout):
  // lanes 0-15: K = 0..7 / 16..23 ; lanes 16-31: K = 8..15 / 24..31
  const int mloc = lane & 15;
  const int kb   = (lane >> 4) << 3;

  issue(0, 0);
  int cur = 0;
  for (int k0 = 0; k0 < KTOT; k0 += BK) {
    const int knext = (k0 + BK < KTOT) ? (k0 + BK) : 0;  // dummy refetch on last iter
    issue(cur ^ 1, knext);
    // async loads complete in order: <=4 outstanding means current buffer landed
    asm volatile("s_wait_asynccnt 0x4" ::: "memory");
    __syncthreads();

    v16bf af[2], bf[4];
#pragma unroll
    for (int mt = 0; mt < 2; ++mt) {
      const bf16_t* p = &As[cur][(wm * 32 + mt * 16 + mloc) * LDSROW + kb];
      v8bf lo = *(const v8bf*)p;
      v8bf hi = *(const v8bf*)(p + 16);
      af[mt] = __builtin_shufflevector(lo, hi, 0,1,2,3,4,5,6,7,8,9,10,11,12,13,14,15);
    }
#pragma unroll
    for (int nt = 0; nt < 4; ++nt) {
      const bf16_t* p = &Bs[cur][(wn * 64 + nt * 16 + mloc) * LDSROW + kb];
      v8bf lo = *(const v8bf*)p;
      v8bf hi = *(const v8bf*)(p + 16);
      bf[nt] = __builtin_shufflevector(lo, hi, 0,1,2,3,4,5,6,7,8,9,10,11,12,13,14,15);
    }
#pragma unroll
    for (int mt = 0; mt < 2; ++mt)
#pragma unroll
      for (int nt = 0; nt < 4; ++nt)
        acc[mt][nt] = __builtin_amdgcn_wmma_f32_16x16x32_bf16(
            false, af[mt], false, bf[nt], (short)0, acc[mt][nt], false, false);

    __syncthreads();   // all reads of cur done before it is refilled next iteration
    cur ^= 1;
  }

  // Epilogue. D layout: VGPR r -> M = r (lanes 0-15) / 8+r (lanes 16-31), N = lane%16.
  const int half8 = (lane >> 4) << 3;
#pragma unroll
  for (int mt = 0; mt < 2; ++mt) {
#pragma unroll
    for (int nt = 0; nt < 4; ++nt) {
      const int n = (int)n0 + wn * 64 + nt * 16 + mloc;
      float bval = 0.0f;
      if constexpr (HAS_BIAS) bval = bias[n];
      const long long mrow = m0 + wm * 32 + mt * 16 + half8;
      if constexpr (OUTF32) {
        float* p = Cf + (long long)blockIdx.z * SCZ + mrow * LDC + n;
#pragma unroll
        for (int r = 0; r < 8; ++r) p[r * LDC] = acc[mt][nt][r] * scale + bval;
      }
      if constexpr (SPLIT) {
        const int which = n / SPLITN;              // compile-time SPLITN -> shifts
        const int nn = n - which * SPLITN;
        bf16_t* p = Cb + (long long)which * PLANE + mrow * SPLITN + nn;
#pragma unroll
        for (int r = 0; r < 8; ++r) p[r * SPLITN] = f2bf(acc[mt][nt][r] * scale + bval);
      }
    }
  }
}

// ---------------- row softmax, fp32 in -> bf16 out (in place, row-private slot) ----------------
__global__ void softmax_rows(const float* __restrict__ S, bf16_t* __restrict__ P,
                             int cols, int pstride) {
  const long long row = blockIdx.x;
  const float* srow = S + row * (long long)cols;
  bf16_t* prow = P + row * (long long)pstride;
  float vals[8];
  float m = -INFINITY;
#pragma unroll
  for (int i = 0; i < 8; ++i) {
    vals[i] = srow[threadIdx.x + i * 256];
    m = fmaxf(m, vals[i]);
  }
  __shared__ float red[256];
  red[threadIdx.x] = m;
  __syncthreads();
  for (int s = 128; s > 0; s >>= 1) {
    if (threadIdx.x < s) red[threadIdx.x] = fmaxf(red[threadIdx.x], red[threadIdx.x + s]);
    __syncthreads();
  }
  m = red[0];
  __syncthreads();
  float sum = 0.0f;
#pragma unroll
  for (int i = 0; i < 8; ++i) {
    vals[i] = __expf(vals[i] - m);
    sum += vals[i];
  }
  red[threadIdx.x] = sum;
  __syncthreads();
  for (int s = 128; s > 0; s >>= 1) {
    if (threadIdx.x < s) red[threadIdx.x] += red[threadIdx.x + s];
    __syncthreads();
  }
  float inv = 1.0f / red[0];
#pragma unroll
  for (int i = 0; i < 8; ++i)
    prow[threadIdx.x + i * 256] = f2bf(vals[i] * inv);
}

// ---------------- bf16 transpose: Vt[b][h][s] = V[b][s][h] ----------------
__global__ void transpose_bf16(const bf16_t* __restrict__ V, bf16_t* __restrict__ Vt,
                               int S, int H) {
  __shared__ bf16_t t[32][33];
  const long long plane = (long long)S * H;
  const bf16_t* Vb = V + (long long)blockIdx.z * plane;
  bf16_t* Vtb = Vt + (long long)blockIdx.z * plane;
  int h0 = blockIdx.x * 32, s0 = blockIdx.y * 32;
  for (int i = threadIdx.y; i < 32; i += 8)
    t[i][threadIdx.x] = Vb[(long long)(s0 + i) * H + h0 + threadIdx.x];
  __syncthreads();
  for (int i = threadIdx.y; i < 32; i += 8)
    Vtb[(long long)(h0 + i) * S + s0 + threadIdx.x] = t[threadIdx.x][i];
}

// ---------------- host orchestration ----------------
extern "C" void kernel_launch(void* const* d_in, const int* in_sizes, int n_in,
                              void* d_out, int out_size, void* d_ws, size_t ws_size,
                              hipStream_t stream) {
  (void)in_sizes; (void)n_in; (void)out_size; (void)ws_size;
  constexpr int Bn = 4, S = 2048, H = 1024;
  constexpr long long Mall = (long long)Bn * S;          // 8192

  const float* X  = (const float*)d_in[0];
  const float* Wq = (const float*)d_in[1];
  const float* bq = (const float*)d_in[2];
  const float* Wk = (const float*)d_in[3];
  const float* bk = (const float*)d_in[4];
  const float* Wv = (const float*)d_in[5];
  const float* bv = (const float*)d_in[6];
  float* out = (float*)d_out;

  char* ws = (char*)d_ws;
  size_t off = 0;
  auto wsalloc = [&](size_t bytes) -> void* {
    void* p = ws + off;
    off = (off + bytes + 255) & ~(size_t)255;
    return p;
  };
  bf16_t* Xb   = (bf16_t*)wsalloc((size_t)Mall * H * 2);
  bf16_t* Wqkv = (bf16_t*)wsalloc((size_t)3 * H * H * 2);
  float*  bqkv = (float*) wsalloc((size_t)3 * H * 4);
  bf16_t* Qb   = (bf16_t*)wsalloc((size_t)Mall * H * 2);  // Q,K,V consecutive planes
  bf16_t* Kb   = (bf16_t*)wsalloc((size_t)Mall * H * 2);
  bf16_t* Vb   = (bf16_t*)wsalloc((size_t)Mall * H * 2);
  bf16_t* Vt   = (bf16_t*)wsalloc((size_t)Mall * H * 2);
  float*  Sc   = (float*) wsalloc((size_t)Bn * S * S * 4); // scores; P bf16 reuses in place
  (void)Kb;

  // 1) precision conversion + weight packing
  cvt_f32_bf16<<<2048, 256, 0, stream>>>((const float4*)X, (ushort4*)Xb, (int)(Mall * H / 4));
  cvt_f32_bf16<<<512, 256, 0, stream>>>((const float4*)Wq, (ushort4*)Wqkv,                      H * H / 4);
  cvt_f32_bf16<<<512, 256, 0, stream>>>((const float4*)Wk, (ushort4*)(Wqkv + (size_t)H * H),    H * H / 4);
  cvt_f32_bf16<<<512, 256, 0, stream>>>((const float4*)Wv, (ushort4*)(Wqkv + (size_t)2 * H * H), H * H / 4);
  hipMemcpyAsync(bqkv,         bq, H * sizeof(float), hipMemcpyDeviceToDevice, stream);
  hipMemcpyAsync(bqkv + H,     bk, H * sizeof(float), hipMemcpyDeviceToDevice, stream);
  hipMemcpyAsync(bqkv + 2 * H, bv, H * sizeof(float), hipMemcpyDeviceToDevice, stream);

  // 2) fused QKV projection: [8192,1024] @ [3072,1024]^T + bias -> bf16 Q/K/V planes
  gemm_bf16_nt<H, H, H, 0, 0,
               true, false, 1, 0,
               true, H, Mall * H>
      <<<dim3(3 * H / BN, Mall / BM, 1), 256, 0, stream>>>(
          Xb, Wqkv, nullptr, Qb, bqkv, 1.0f);

  // 3) V -> V^T
  transpose_bf16<<<dim3(H / 32, S / 32, Bn), dim3(32, 8), 0, stream>>>(Vb, Vt, S, H);

  // 4) scores = Q @ K^T / sqrt(H), fp32 per batch (lives in 192MB L2)
  gemm_bf16_nt<H, H, H, (long long)S * H, (long long)S * H,
               false, true, S, (long long)S * S,
               false, 1, 1>
      <<<dim3(S / BN, S / BM, Bn), 256, 0, stream>>>(
          Qb, Qb + Mall * H /*K plane*/, Sc, nullptr, nullptr, 0.03125f);

  // 5) row softmax; bf16 P written in place (row stride 2*S bf16 = row's own 8KB slot)
  softmax_rows<<<Bn * S, 256, 0, stream>>>(Sc, (bf16_t*)Sc, S, 2 * S);

  // 6) out = P @ V : A = P [2048 x 2048] (lda=4096), B = Vt [1024 x 2048]
  gemm_bf16_nt<2 * S, S, S, (long long)S * (2 * S), (long long)H * S,
               false, true, H, (long long)S * H,
               false, 1, 1>
      <<<dim3(H / BN, S / BM, Bn), 256, 0, stream>>>(
          (const bf16_t*)Sc, Vt, out, nullptr, nullptr, 1.0f);
}
